// Seq2SeqModel_10411000725956
// MI455X (gfx1250) — compile-verified, matrix operand order checked by reading, not run on hardware
//
#include <hip/hip_runtime.h>
#include <math.h>

#define Hdim 1024
#define Bsz  64
#define Slen 64
#define Tlen 64
#define VTsz 32000

#define KC   64      // K-chunk depth (two 32-deep WMMA sub-chunks)
#define LSTR 72      // LDS row stride in ushorts (144 B: 16B-aligned, bank-dephased)

typedef __attribute__((ext_vector_type(16))) __bf16          v16bf;
typedef __attribute__((ext_vector_type(16))) unsigned short  vus16;
typedef __attribute__((ext_vector_type(8)))  float           v8f;
typedef __attribute__((ext_vector_type(4)))  float           f32x4;
typedef __attribute__((ext_vector_type(4)))  unsigned int    u32x4;

__device__ __forceinline__ unsigned short f32_to_bf16(float f) {
  union { float f; unsigned u; } v; v.f = f;
  unsigned u = v.u;
  u += 0x7FFFu + ((u >> 16) & 1u);   // round-to-nearest-even
  return (unsigned short)(u >> 16);
}

__device__ __forceinline__ float sigmoidf_(float x) { return 1.0f / (1.0f + __expf(-x)); }

// ---------------------------------------------------------------------------
// Skinny GEMM: C[64,N] = act( A[64,K0+K1] * B[(K0+K1),N] + bias )
// A first K0 cols: gathered embedding rows (tok != nullptr) or A0 rows.
// A last  K1 cols: A1 rows (recurrent h). B split at K0 into B0/B1.
// Block = 128 threads (4 waves). Block tile = 64(M) x 64(N), K chunk = 64.
// Wave w: rows 16w..16w+15, four 16x16 N-tiles; 8 WMMAs per barrier pair.
// ---------------------------------------------------------------------------
__global__ __launch_bounds__(128)
void gemm_bf16_wmma(const float* __restrict__ A0,
                    const int*   __restrict__ tok, int tokStride,
                    const float* __restrict__ emb,
                    const float* __restrict__ A1,
                    const float* __restrict__ B0,
                    const float* __restrict__ B1,
                    const float* __restrict__ bias0,
                    const float* __restrict__ bias1,
                    float* __restrict__ C,
                    int N, int K0, int K1, int act)
{
  __shared__ unsigned short As[64 * LSTR];   // [m][k]  9 KB
  __shared__ unsigned short Bs[64 * LSTR];   // [n][k]  9 KB (transposed)

  const int tid  = threadIdx.x;
  const int lane = tid & 31;
  const int wv   = tid >> 5;
  const int n0   = blockIdx.x * 64;
  const int K    = K0 + K1;

  // ---- per-thread staging coordinates (all in registers, no LDS tables) ----
  const int rA = tid >> 1;               // A row this thread stages
  const int hA = (tid & 1) * 32;         // k-half within chunk
  const int kB = tid >> 1;               // B k-row this thread stages
  const int hB = (tid & 1) * 32;         // n-half within 64-wide block tile

  const float* rowA0 = tok ? (emb + (size_t)tok[(size_t)rA * tokStride] * K0)
                           : (A0 ? A0 + (size_t)rA * K0 : (const float*)0);
  const float* rowA1 = A1 ? (A1 + (size_t)rA * K1) : (const float*)0;

  v8f acc0 = {}, acc1 = {}, acc2 = {}, acc3 = {};

  for (int kb = 0; kb < K; kb += KC) {
    const int seg  = (kb < K0) ? 0 : 1;                 // uniform per chunk
    const int kOff = (seg == 0) ? kb : (kb - K0);
    const float* __restrict__ Bbase =
        (seg == 0) ? (B0 + (size_t)kOff * N) : (B1 + (size_t)kOff * N);

    // ---- stage A: thread loads 32 contiguous f32 of its row (8x b128) ----
    {
      const float* asrc = ((seg == 0) ? rowA0 : rowA1) + kOff + hA;
      unsigned int pk[16];
#pragma unroll
      for (int q = 0; q < 8; ++q) {
        f32x4 x = *(const f32x4*)(asrc + q * 4);
        pk[2 * q + 0] = (unsigned)f32_to_bf16(x[0]) | ((unsigned)f32_to_bf16(x[1]) << 16);
        pk[2 * q + 1] = (unsigned)f32_to_bf16(x[2]) | ((unsigned)f32_to_bf16(x[3]) << 16);
      }
      u32x4* dst = (u32x4*)&As[rA * LSTR + hA];
#pragma unroll
      for (int q = 0; q < 4; ++q) {
        u32x4 w = { pk[4 * q], pk[4 * q + 1], pk[4 * q + 2], pk[4 * q + 3] };
        dst[q] = w;
      }
    }
    // ---- stage B: thread loads 32 contiguous n of one k-row, scatter-T ----
    {
      const float* bsrc = Bbase + (size_t)kB * N + n0 + hB;
#pragma unroll
      for (int q = 0; q < 8; ++q) {
        f32x4 x = *(const f32x4*)(bsrc + q * 4);
#pragma unroll
        for (int e = 0; e < 4; ++e) {
          int nn = hB + q * 4 + e;
          Bs[nn * LSTR + kB] = f32_to_bf16(x[e]);
        }
      }
    }
    // prefetch next weight chunk (global_prefetch_b8)
    int kn = kb + KC;
    if (kn < K) {
      const float* pb = (kn < K0) ? (B0 + (size_t)kn * N + n0)
                                  : (B1 + (size_t)(kn - K0) * N + n0);
      __builtin_prefetch(pb, 0, 1);
    }
    __syncthreads();

    // ---- compute: 2 sub-chunks x 4 N-tiles = 8 WMMAs ----
    const int m = lane & 15;
    const int g = lane >> 4;
#pragma unroll
    for (int sc = 0; sc < 2; ++sc) {
      union { vus16 u; v16bf b; } af, bf0, bf1, bf2, bf3;
#pragma unroll
      for (int e = 0; e < 16; ++e) {
        int k = sc * 32 + 8 * g + (e & 7) + ((e >> 3) << 4);
        af.u[e]  = As[(wv * 16 + m) * LSTR + k];
        bf0.u[e] = Bs[(0 * 16 + m) * LSTR + k];
        bf1.u[e] = Bs[(1 * 16 + m) * LSTR + k];
        bf2.u[e] = Bs[(2 * 16 + m) * LSTR + k];
        bf3.u[e] = Bs[(3 * 16 + m) * LSTR + k];
      }
      acc0 = __builtin_amdgcn_wmma_f32_16x16x32_bf16(false, af.b, false, bf0.b, (short)0, acc0, false, false);
      acc1 = __builtin_amdgcn_wmma_f32_16x16x32_bf16(false, af.b, false, bf1.b, (short)0, acc1, false, false);
      acc2 = __builtin_amdgcn_wmma_f32_16x16x32_bf16(false, af.b, false, bf2.b, (short)0, acc2, false, false);
      acc3 = __builtin_amdgcn_wmma_f32_16x16x32_bf16(false, af.b, false, bf3.b, (short)0, acc3, false, false);
    }
    __syncthreads();
  }

  // ---- epilogue: C layout VGPR r -> M = r + 8*(lane>=16), N = lane%16 ----
  const int colL = lane & 15;
  const int hiM  = (lane >> 4) * 8;
  v8f accs[4] = { acc0, acc1, acc2, acc3 };
#pragma unroll
  for (int j = 0; j < 4; ++j) {
#pragma unroll
    for (int r = 0; r < 8; ++r) {
      int mrow = wv * 16 + hiM + r;
      int ncol = n0 + j * 16 + colL;
      float v = accs[j][r];
      if (bias0) v += bias0[ncol];
      if (bias1) v += bias1[ncol];
      if (act)   v = tanhf(v);
      C[(size_t)mrow * N + ncol] = v;
    }
  }
}

// ---------------------------------------------------------------------------
__global__ void zero_init(float* c, float* h, float* out) {
  int idx = blockIdx.x * blockDim.x + threadIdx.x;
  if (idx < Bsz * Hdim) { c[idx] = 0.f; h[idx] = 0.f; }
  if (idx == 0) out[0] = 0.f;
}

__global__ void lstm_pointwise(const float* __restrict__ gates,
                               float* __restrict__ c, float* __restrict__ h,
                               float* __restrict__ hist_c, float* __restrict__ hist_h,
                               int sidx)
{
  int idx = blockIdx.x * blockDim.x + threadIdx.x;
  if (idx >= Bsz * Hdim) return;
  int b = idx >> 10, j = idx & 1023;
  const float* g = gates + (size_t)b * 4 * Hdim;
  float gi = g[j], gf = g[Hdim + j], gg = g[2 * Hdim + j], go = g[3 * Hdim + j];
  float cn = sigmoidf_(gf) * c[idx] + sigmoidf_(gi) * tanhf(gg);
  float hn = sigmoidf_(go) * tanhf(cn);
  c[idx] = cn; h[idx] = hn;
  if (hist_c) {
    hist_c[(size_t)sidx * Bsz * Hdim + idx] = cn;
    hist_h[(size_t)sidx * Bsz * Hdim + idx] = hn;
  }
}

__global__ void dec_init(const int* __restrict__ src_lens,
                         const float* __restrict__ hist_c,
                         const float* __restrict__ hist_h,
                         float* __restrict__ c, float* __restrict__ h)
{
  int idx = blockIdx.x * blockDim.x + threadIdx.x;
  if (idx >= Bsz * Hdim) return;
  int b = idx >> 10;
  int s = src_lens[b] - 1;
  s = s < 0 ? 0 : (s > Slen - 1 ? Slen - 1 : s);
  c[idx] = hist_c[(size_t)s * Bsz * Hdim + idx];
  h[idx] = hist_h[(size_t)s * Bsz * Hdim + idx];
}

__global__ void attn_scores(const float* __restrict__ hist_h,
                            const float* __restrict__ h,
                            float* __restrict__ dotb)
{
  int b = blockIdx.x >> 6, s = blockIdx.x & 63;
  __shared__ float red[256];
  float p = 0.f;
  for (int j = threadIdx.x; j < Hdim; j += 256)
    p += hist_h[(size_t)s * Bsz * Hdim + (size_t)b * Hdim + j] * h[(size_t)b * Hdim + j];
  red[threadIdx.x] = p; __syncthreads();
  for (int off = 128; off > 0; off >>= 1) {
    if (threadIdx.x < off) red[threadIdx.x] += red[threadIdx.x + off];
    __syncthreads();
  }
  if (threadIdx.x == 0) dotb[b * Slen + s] = red[0];
}

__global__ void softmax_ctx(const float* __restrict__ dotb,
                            const float* __restrict__ hist_h,
                            float* __restrict__ ctx)
{
  int b = blockIdx.x;
  __shared__ float attn[Slen];
  if (threadIdx.x == 0) {
    float mx = -1e30f;
    for (int s = 0; s < Slen; ++s) mx = fmaxf(mx, dotb[b * Slen + s]);
    float sm = 0.f;
    for (int s = 0; s < Slen; ++s) { float e = __expf(dotb[b * Slen + s] - mx); attn[s] = e; sm += e; }
    float inv = 1.f / sm;
    for (int s = 0; s < Slen; ++s) attn[s] *= inv;
  }
  __syncthreads();
  for (int j = threadIdx.x; j < Hdim; j += 256) {
    float a = 0.f;
    for (int s = 0; s < Slen; ++s)
      a += attn[s] * hist_h[(size_t)s * Bsz * Hdim + (size_t)b * Hdim + j];
    ctx[(size_t)b * Hdim + j] = a;
  }
}

__global__ void nll_rows(const float* __restrict__ logits,
                         const int* __restrict__ tgt_col, int tgtStride,
                         float* __restrict__ nll_out)
{
  int b = blockIdx.x;
  const float* row = logits + (size_t)b * VTsz;
  __shared__ float red[256];
  float mx = -1e30f;
  for (int j = threadIdx.x; j < VTsz; j += 256) mx = fmaxf(mx, row[j]);
  red[threadIdx.x] = mx; __syncthreads();
  for (int off = 128; off > 0; off >>= 1) {
    if (threadIdx.x < off) red[threadIdx.x] = fmaxf(red[threadIdx.x], red[threadIdx.x + off]);
    __syncthreads();
  }
  mx = red[0]; __syncthreads();
  float sm = 0.f;
  for (int j = threadIdx.x; j < VTsz; j += 256) sm += __expf(row[j] - mx);
  red[threadIdx.x] = sm; __syncthreads();
  for (int off = 128; off > 0; off >>= 1) {
    if (threadIdx.x < off) red[threadIdx.x] += red[threadIdx.x + off];
    __syncthreads();
  }
  if (threadIdx.x == 0) {
    int tgt = tgt_col[(size_t)b * tgtStride];
    nll_out[b] = (tgt != 0) ? -(row[tgt] - mx - __logf(red[0])) : 0.f;
  }
}

__global__ void final_sum(const float* __restrict__ nll, int n, float* __restrict__ out) {
  __shared__ float red[256];
  float s = 0.f;
  for (int i = threadIdx.x; i < n; i += 256) s += nll[i];
  red[threadIdx.x] = s; __syncthreads();
  for (int off = 128; off > 0; off >>= 1) {
    if (threadIdx.x < off) red[threadIdx.x] += red[threadIdx.x + off];
    __syncthreads();
  }
  if (threadIdx.x == 0) out[0] = red[0];
}

// ---------------------------------------------------------------------------
extern "C" void kernel_launch(void* const* d_in, const int* in_sizes, int n_in,
                              void* d_out, int out_size, void* d_ws, size_t ws_size,
                              hipStream_t stream)
{
  (void)in_sizes; (void)n_in; (void)out_size; (void)ws_size;

  const int*   src_tokens = (const int*)  d_in[0];
  const int*   tgt_tokens = (const int*)  d_in[1];
  const int*   src_lens   = (const int*)  d_in[2];
  const float* emb_src    = (const float*)d_in[3];
  const float* emb_tgt    = (const float*)d_in[4];
  const float* enc_Wi     = (const float*)d_in[5];
  const float* enc_bi     = (const float*)d_in[6];
  const float* enc_Wh     = (const float*)d_in[7];
  const float* enc_bh     = (const float*)d_in[8];
  const float* dec_Wi     = (const float*)d_in[9];
  const float* dec_bi     = (const float*)d_in[10];
  const float* dec_Wh     = (const float*)d_in[11];
  const float* dec_bh     = (const float*)d_in[12];
  const float* att_W      = (const float*)d_in[13];
  const float* att_b      = (const float*)d_in[14];
  const float* out_W      = (const float*)d_in[15];
  const float* out_b      = (const float*)d_in[16];
  float* out = (float*)d_out;

  float* p = (float*)d_ws;
  float* enc_c_hist = p; p += (size_t)Slen * Bsz * Hdim;   // [S,B,H]
  float* enc_h_hist = p; p += (size_t)Slen * Bsz * Hdim;   // [S,B,H]
  float* c      = p; p += (size_t)Bsz * Hdim;
  float* h      = p; p += (size_t)Bsz * Hdim;
  float* gates  = p; p += (size_t)Bsz * 4 * Hdim;
  float* dotb   = p; p += (size_t)Bsz * Slen;
  float* ctx    = p; p += (size_t)Bsz * Hdim;
  float* hid    = p; p += (size_t)Bsz * Hdim;
  float* logits = p; p += (size_t)Bsz * VTsz;
  float* nllbuf = p; p += (size_t)(Tlen - 1) * Bsz;

  const int pwGrid = (Bsz * Hdim + 255) / 256;

  zero_init<<<pwGrid, 256, 0, stream>>>(c, h, out);

  // ---------------- encoder ----------------
  for (int s = 0; s < Slen; ++s) {
    gemm_bf16_wmma<<<(4 * Hdim) / 64, 128, 0, stream>>>(
        nullptr, src_tokens + s, Slen, emb_src,
        h, enc_Wi, enc_Wh, enc_bi, enc_bh,
        gates, 4 * Hdim, Hdim, Hdim, 0);
    lstm_pointwise<<<pwGrid, 256, 0, stream>>>(gates, c, h, enc_c_hist, enc_h_hist, s);
  }
  dec_init<<<pwGrid, 256, 0, stream>>>(src_lens, enc_c_hist, enc_h_hist, c, h);

  // ---------------- decoder ----------------
  for (int t = 0; t < Tlen - 1; ++t) {
    gemm_bf16_wmma<<<(4 * Hdim) / 64, 128, 0, stream>>>(
        nullptr, tgt_tokens + t, Tlen, emb_tgt,
        h, dec_Wi, dec_Wh, dec_bi, dec_bh,
        gates, 4 * Hdim, Hdim, Hdim, 0);
    lstm_pointwise<<<pwGrid, 256, 0, stream>>>(gates, c, h, nullptr, nullptr, 0);

    attn_scores<<<Bsz * Slen, 256, 0, stream>>>(enc_h_hist, h, dotb);
    softmax_ctx<<<Bsz, 256, 0, stream>>>(dotb, enc_h_hist, ctx);

    // hid = tanh([ctx|h] @ att_W + att_b)
    gemm_bf16_wmma<<<Hdim / 64, 128, 0, stream>>>(
        ctx, nullptr, 0, nullptr,
        h, att_W, att_W + (size_t)Hdim * Hdim, att_b, nullptr,
        hid, Hdim, Hdim, Hdim, 1);

    // logits = hid @ out_W + out_b
    gemm_bf16_wmma<<<VTsz / 64, 128, 0, stream>>>(
        hid, nullptr, 0, nullptr,
        nullptr, out_W, nullptr, out_b, nullptr,
        logits, VTsz, Hdim, 0, 0);

    nll_rows<<<Bsz, 256, 0, stream>>>(logits, tgt_tokens + (t + 1), Tlen, nllbuf + t * Bsz);
  }

  final_sum<<<1, 256, 0, stream>>>(nllbuf, (Tlen - 1) * Bsz, out);
}
